// ShuffleNet_69965017252068
// MI455X (gfx1250) — matrix operand order repaired and checked by reading, not compiled, not run on hardware
//
#include <hip/hip_runtime.h>
#include <hip/hip_bf16.h>
#include <math.h>

typedef __attribute__((ext_vector_type(16))) __bf16 v16bf;
typedef __attribute__((ext_vector_type(8)))  __bf16 v8bf;
typedef __attribute__((ext_vector_type(8)))  float  v8f;

#define HIDDEN 1024
#define NB 16           // number of blocks
#define BS 64           // block size (GEMM1 K per block)
#define IBS 256         // intermediate block size
#define TOKS_PER_WG 32
#define XPAD 8          // pad (halfs) on x rows -> banks spread for b128 loads
#define HPAD 8          // pad (halfs) on h2 rows
#define XROW (HIDDEN + XPAD)
#define HROW (IBS + HPAD)
#define XS_ELEMS (TOKS_PER_WG * XROW)            // 33024 (66 KB) -> offset 0
#define H2_ELEMS_PER_WAVE (TOKS_PER_WG * HROW)   // 8448 (16.5 KB)
#define W1F_ELEMS (NB * IBS * BS)        // 262144
#define W2F_ELEMS (NB * BS * IBS)        // 262144

// ---------------------------------------------------------------------------
// D = A*B + C, 16x16x32 bf16 -> f32
// ---------------------------------------------------------------------------
static __device__ __forceinline__ v8f wmma_bf16(v16bf a, v16bf b, v8f c) {
  return __builtin_amdgcn_wmma_f32_16x16x32_bf16(false, a, false, b,
                                                 (short)0, c, false, false);
}

static __device__ __forceinline__ v8f splat8(float v) {
  v8f r;
#pragma unroll
  for (int i = 0; i < 8; ++i) r[i] = v;
  return r;
}

// A-matrix 16x32 bf16 fragment from an LDS row (row-major, K contiguous).
// Lanes 0-15 (M=lane):  elems 0-7 = K+0..7,  elems 8-15 = K+16..23
// Lanes 16-31 (M=lane-16): elems 0-7 = K+8..15, elems 8-15 = K+24..31
// hsel (lane>=16) is folded into the caller's base pointer.
static __device__ __forceinline__ v16bf load_a_frag(const __bf16* rowk_h) {
  union { v16bf v; v8bf h[2]; } u;
  u.h[0] = *(const v8bf*)(rowk_h);
  u.h[1] = *(const v8bf*)(rowk_h + 16);
  return u.v;
}

// Branch-free erf (Abramowitz-Stegun 7.1.26, |err| <= 1.5e-7), using the
// hardware-approx rcp and exp2 TRANS ops instead of libm's branchy erff.
static __device__ __forceinline__ float erf_fast(float x) {
  float z = fabsf(x);
  float t = __builtin_amdgcn_rcpf(fmaf(0.3275911f, z, 1.0f));
  float p = fmaf(t, 1.061405429f, -1.453152027f);
  p = fmaf(t, p, 1.421413741f);
  p = fmaf(t, p, -0.284496736f);
  p = fmaf(t, p, 0.254829592f);
  p = p * t;
  float e = __builtin_amdgcn_exp2f(-z * z * 1.4426950408889634f);
  float r = fmaf(-p, e, 1.0f);
  return copysignf(r, x);
}

static __device__ __forceinline__ float gelu_fast(float v) {
  return 0.5f * v * (1.0f + erf_fast(v * 0.70710678118654752f));
}

// ---------------------------------------------------------------------------
// Prep: convert w1/w2 fp32 -> bf16 and shuffle into WMMA B-fragment register
// order. A 32x16 (KxN) bf16 B fragment is 512 elements; element (K,N) lives
// at lane = N + 16*((K>>3)&1), e = (K&7) + 8*(K>>4)  ->  flat index lane*16 + e, so each lane
// reads its 16 halfs with one contiguous 32-byte load.
//
// w1 frags keyed by (np, n, s):  value = w1[n][np*16 + N][s*32 + K]
// w2 frags keyed by (np, ot, kk): value = w2[np][ot*16 + N][kk*32 + K]
// ---------------------------------------------------------------------------
__global__ void prep_weights(const float* __restrict__ w1,
                             const float* __restrict__ w2,
                             __bf16* __restrict__ w1f,
                             __bf16* __restrict__ w2f) {
  int tid = blockIdx.x * blockDim.x + threadIdx.x;   // 0 .. 2*262144-1
  int sel = tid >> 18;
  int t   = tid & (W1F_ELEMS - 1);
  int f    = t >> 9;          // fragment id
  int r    = t & 511;
  int lane = r >> 4;
  int e    = r & 15;
  int N    = lane & 15;
  int K    = (e & 7) + ((lane >> 4) << 3) + ((e >> 3) << 4);
  int np   = f >> 5;
  int rem  = f & 31;
  if (sel == 0) {
    int n = rem >> 1, s = rem & 1;
    w1f[t] = (__bf16)w1[(n * IBS + np * 16 + N) * BS + s * 32 + K];
  } else {
    int ot = rem >> 3, kk = rem & 7;
    w2f[t] = (__bf16)w2[(np * BS + ot * 16 + N) * IBS + kk * 32 + K];
  }
}

// ---------------------------------------------------------------------------
// Fused: x -> GEMM1(block-diag) -> shuffle+bias+GeLU (in LDS) -> GEMM2 -> out
// 8 waves, 32 tokens per workgroup; wave w owns second-stage blocks 2w, 2w+1.
// Single merged LDS array: xs at offset 0 so all its DS offsets fit in the
// 16-bit immediate; one materialized base per wave for the h2 slab.
// Biases are pre-loaded into the WMMA C accumulators (bias is constant over
// the C-tile's M dimension), so epilogues are pure GeLU/store.
// ---------------------------------------------------------------------------
__global__ __launch_bounds__(256, 2) void shufflenet_fused(
    const float* __restrict__ x, const __bf16* __restrict__ w1f,
    const __bf16* __restrict__ w2f, const float* __restrict__ b1,
    const float* __restrict__ b2, float* __restrict__ out) {
  __shared__ __align__(16) __bf16 smem[XS_ELEMS + 8 * H2_ELEMS_PER_WAVE];
  __bf16* const xs = smem;                       // [32][XROW], offset 0

  const int tok0 = blockIdx.x * TOKS_PER_WG;

  // Cooperative, coalesced x load: fp32 -> bf16 into LDS.
  const float4* xin = (const float4*)(x + (size_t)tok0 * HIDDEN);
  for (int i = threadIdx.x; i < TOKS_PER_WG * HIDDEN / 4; i += 256) {
    float4 v = xin[i];
    int row = i >> 8;            // 256 float4 per row
    int col = (i & 255) << 2;
    union { __bf16 h[4]; uint2 u; } p;
    p.h[0] = (__bf16)v.x; p.h[1] = (__bf16)v.y;
    p.h[2] = (__bf16)v.z; p.h[3] = (__bf16)v.w;
    *(uint2*)&xs[row * XROW + col] = p.u;
  }
  __syncthreads();

  const int wave = threadIdx.x >> 5;
  const int lane = threadIdx.x & 31;
  const int hsel = lane >> 4;      // which 16-lane half
  const int tcol = lane & 15;      // N (and C-tile column / A-row select)

  // Per-lane base pointers; everything loop-varying becomes a DS immediate.
  const __bf16* xr0 = xs + tcol * XROW + 8 * hsel;          // A rows 0..15
  const __bf16* xr1 = xs + (16 + tcol) * XROW + 8 * hsel;   // A rows 16..31
  __bf16* const h2base = smem + XS_ELEMS + wave * H2_ELEMS_PER_WAVE;
  const __bf16* h2r0 = h2base + tcol * HROW + 8 * hsel;
  const __bf16* h2r1 = h2base + (16 + tcol) * HROW + 8 * hsel;
  __bf16* hst0 = h2base + (8 * hsel) * HROW + tcol * 16;    // C scatter
  __bf16* hst1 = h2base + (16 + 8 * hsel) * HROW + tcol * 16;

  float* const out_u = out + (size_t)tok0 * HIDDEN;         // uniform base
  const unsigned orow = 8 * hsel * HIDDEN;                  // divergent part

#pragma unroll 1
  for (int ii = 0; ii < 2; ++ii) {
    const int np = wave * 2 + ii;
    // Uniform weight bases + 32-bit divergent lane offset -> GVS addressing.
    const __bf16* const w1b = w1f + ((np * NB * 2) << 9);
    const __bf16* const w2b = w2f + ((np * 32) << 9);
    const unsigned lo = (unsigned)lane * 16u;
    const float* b1base = b1 + np * IBS + tcol * 16;

    // -------- Stage 1: h tiles for blocks (n, n+1), paired so the two GeLU
    // results at shuffled indices t*16+n, t*16+n+1 pack into one b32 store.
#pragma unroll 1
    for (int n = 0; n < NB; n += 2) {
      const v8f c0a = splat8(b1base[n]);      // bias pre-loaded into C
      const v8f c0b = splat8(b1base[n + 1]);
      v8f acc0a = c0a, acc1a = c0a;           // block n,   rows 0..15 / 16..31
      v8f acc0b = c0b, acc1b = c0b;           // block n+1
#pragma unroll
      for (int s = 0; s < 2; ++s) {
        const v16bf bfa = *(const v16bf*)(w1b + (lo + ((n * 2 + s) << 9)));
        const v16bf bfb = *(const v16bf*)(w1b + (lo + ((n * 2 + 2 + s) << 9)));
        const int ka = n * BS + s * 32;
        const int kb = ka + BS;
        v16bf a0a = load_a_frag(xr0 + ka);
        v16bf a1a = load_a_frag(xr1 + ka);
        v16bf a0b = load_a_frag(xr0 + kb);
        v16bf a1b = load_a_frag(xr1 + kb);
        acc0a = wmma_bf16(a0a, bfa, acc0a);
        acc1a = wmma_bf16(a1a, bfa, acc1a);
        acc0b = wmma_bf16(a0b, bfb, acc0b);
        acc1b = wmma_bf16(a1b, bfb, acc1b);
      }
#pragma unroll
      for (int r = 0; r < 8; ++r) {
        union { __bf16 h[2]; unsigned u; } p0, p1;
        p0.h[0] = (__bf16)gelu_fast(acc0a[r]);
        p0.h[1] = (__bf16)gelu_fast(acc0b[r]);
        p1.h[0] = (__bf16)gelu_fast(acc1a[r]);
        p1.h[1] = (__bf16)gelu_fast(acc1b[r]);
        *(unsigned*)(hst0 + r * HROW + n) = p0.u;
        *(unsigned*)(hst1 + r * HROW + n) = p1.u;
      }
    }
    asm volatile("s_wait_dscnt 0x0" ::: "memory");  // LDS RAW fence (per-wave)

    // -------- Stage 2: y[., np*64 + o'].  kk outer / ot inner so each A
    // fragment loaded from LDS feeds 4 WMMAs (acc[4][2] stays in VGPRs).
    v8f acc[4][2];
#pragma unroll
    for (int ot = 0; ot < 4; ++ot) {
      const v8f c = splat8(b2[np * BS + ot * 16 + tcol]);   // bias in C
      acc[ot][0] = c;
      acc[ot][1] = c;
    }
#pragma unroll 2
    for (int kk = 0; kk < 8; ++kk) {
      v16bf a0 = load_a_frag(h2r0 + kk * 32);
      v16bf a1 = load_a_frag(h2r1 + kk * 32);
#pragma unroll
      for (int ot = 0; ot < 4; ++ot) {
        const v16bf bf = *(const v16bf*)(w2b + (lo + ((ot * 8 + kk) << 9)));
        acc[ot][0] = wmma_bf16(a0, bf, acc[ot][0]);
        acc[ot][1] = wmma_bf16(a1, bf, acc[ot][1]);
      }
    }
#pragma unroll
    for (int ot = 0; ot < 4; ++ot) {
      const unsigned ocol = np * BS + ot * 16 + tcol;
#pragma unroll
      for (int r = 0; r < 8; ++r) {
        out_u[orow + r * HIDDEN + ocol]        = acc[ot][0][r];
        out_u[orow + (16 + r) * HIDDEN + ocol] = acc[ot][1][r];
      }
    }
  }
}

extern "C" void kernel_launch(void* const* d_in, const int* in_sizes, int n_in,
                              void* d_out, int out_size, void* d_ws, size_t ws_size,
                              hipStream_t stream) {
  const float* x  = (const float*)d_in[0];
  const float* w1 = (const float*)d_in[1];
  const float* b1 = (const float*)d_in[2];
  const float* w2 = (const float*)d_in[3];
  const float* b2 = (const float*)d_in[4];
  float* out = (float*)d_out;

  __bf16* w1f = (__bf16*)d_ws;
  __bf16* w2f = w1f + W1F_ELEMS;

  const int tokens = in_sizes[0] / HIDDEN;  // 8192

  prep_weights<<<(W1F_ELEMS + W2F_ELEMS) / 256, 256, 0, stream>>>(w1, w2, w1f, w2f);
  shufflenet_fused<<<tokens / TOKS_PER_WG, 256, 0, stream>>>(x, w1f, w2f, b1,
                                                             b2, out);
}
